// Attention_11390253269691
// MI455X (gfx1250) — matrix operand order repaired and checked by reading, not compiled, not run on hardware
//
#include <hip/hip_runtime.h>

// ---------------------------------------------------------------------------
// Llama-style attention block for MI455X (gfx1250, wave32, WMMA, TDM).
// fp32->bf16 once, all GEMMs on v_wmma_f32_16x16x32_bf16 (fp32 accum),
// GQA flash attention with 64-wide key blocks and online softmax (row-sum
// via an all-ones WMMA -> no cross-lane sum reduction), V pre-transposed so
// P@V B-fragments are contiguous loads. Projection GEMMs stage 64x64 W tiles
// into LDS with the Tensor Data Mover (tensor_load_to_lds + s_wait_tensorcnt),
// double-buffered, shared by all 4 waves of the workgroup.
// ---------------------------------------------------------------------------

typedef __bf16 bf16;
typedef bf16  bf16x8  __attribute__((ext_vector_type(8)));
typedef bf16  bf16x16 __attribute__((ext_vector_type(16)));
typedef float v8f     __attribute__((ext_vector_type(8)));
typedef unsigned int u32x4 __attribute__((ext_vector_type(4)));
typedef int          i32x4 __attribute__((ext_vector_type(4)));
typedef int          i32x8 __attribute__((ext_vector_type(8)));

#define DEVFN __device__ __forceinline__

#if defined(__has_builtin)
#if __has_builtin(__builtin_amdgcn_tensor_load_to_lds)
#define HAVE_TDM 1
#else
#define HAVE_TDM 0
#endif
#else
#define HAVE_TDM 0
#endif

#if HAVE_TDM
#warning "gfx1250: TDM path ENABLED (tensor_load_to_lds builtin found)"
#else
#warning "gfx1250: TDM builtin NOT found - using cooperative LDS fallback"
#endif

// problem constants
constexpr int BATCH = 2;
constexpr int SEQ   = 2048;
constexpr int HID   = 2048;
constexpr int NH    = 16;
constexpr int NKV   = 4;
constexpr int HD    = 128;

DEVFN v8f wmma_bf16(bf16x16 a, bf16x16 b, v8f c) {
  return __builtin_amdgcn_wmma_f32_16x16x32_bf16(
      false, a, false, b, (short)0, c, false, false);
}

// A-fragment (16x32 bf16): lane row = lane&15; lanes<16 carry K {0..15},
// lanes>=16 carry K {16..31}. Two contiguous 16B runs per lane.
DEVFN bf16x16 load_afrag(const bf16* p /* row base + abase */) {
  bf16x8 lo = *(const bf16x8*)p;
  bf16x8 hi = *(const bf16x8*)(p + 8);
  return __builtin_shufflevector(lo, hi, 0,1,2,3,4,5,6,7,8,9,10,11,12,13,14,15);
}

DEVFN void wait_tensorcnt0() {
#if HAVE_TDM
#if __has_builtin(__builtin_amdgcn_s_wait_tensorcnt)
  __builtin_amdgcn_s_wait_tensorcnt(0);
#else
  asm volatile("s_wait_tensorcnt 0" ::: "memory");
#endif
#endif
}

#if HAVE_TDM
// One 2-D TDM descriptor: tile (tile_rows x tile_k elems, 2B each) from a
// row-major tensor with row stride `stride_elems`, packed row-major into LDS.
DEVFN void tdm_load_tile(unsigned lds_addr, const bf16* gaddr,
                         unsigned tile_k, unsigned tile_rows,
                         unsigned tens_k, unsigned tens_rows,
                         unsigned stride_elems) {
  unsigned long long ga = (unsigned long long)gaddr;
  u32x4 g0;
  g0.x = 1u;                                  // count=1, user descriptor
  g0.y = lds_addr;                            // LDS byte address
  g0.z = (unsigned)ga;                        // global_addr[31:0]
  g0.w = (unsigned)(ga >> 32) | 0x80000000u;  // global_addr[56:32] | type=2
  i32x8 g1;
  g1[0] = 0x00010000;                         // data_size = 2 bytes
  g1[1] = (int)((tens_k & 0xFFFFu) << 16);                         // dim0 lo16
  g1[2] = (int)(((tens_k >> 16) & 0xFFFFu) | ((tens_rows & 0xFFFFu) << 16));
  g1[3] = (int)(((tens_rows >> 16) & 0xFFFFu) | (tile_k << 16));   // tile_dim0
  g1[4] = (int)tile_rows;                     // tile_dim1 (tile_dim2 = 0)
  g1[5] = (int)stride_elems;                  // tensor_dim0_stride low 32
  g1[6] = 0;
  g1[7] = 0;
  i32x4 z4 = (i32x4)0;
#if __clang_major__ >= 23
  i32x8 z8 = (i32x8)0;
  __builtin_amdgcn_tensor_load_to_lds(g0, g1, z4, z4, z8, 0);
#else
  __builtin_amdgcn_tensor_load_to_lds(g0, g1, z4, z4, 0);
#endif
}
#endif

// ---------------------------------------------------------------------------
// fp32 -> bf16 convert
// ---------------------------------------------------------------------------
__global__ void cvt_f32_bf16(const float* __restrict__ in,
                             bf16* __restrict__ out, size_t n) {
  size_t i = (size_t)blockIdx.x * blockDim.x + threadIdx.x;
  size_t stride = (size_t)gridDim.x * blockDim.x;
  for (; i < n; i += stride) out[i] = (bf16)in[i];
}

// ---------------------------------------------------------------------------
// GEMM: Y[M,N] = X[M,K] @ W[N,K]^T  (bf16 in, fp32 acc). 4 waves, 64x64 tile.
// W tile (64x64, 8KB) staged into LDS by one TDM descriptor, double-buffered;
// two 32-K WMMA steps per barrier. A-fragments come straight from global.
// MODE 0: bf16 [B, heads, S, hd] | MODE 1: bf16 [B, heads, hd, S] (V^T)
// MODE 2: fp32 row-major [M, N]
// ---------------------------------------------------------------------------
template <int MODE>
__global__ void __launch_bounds__(128)
gemm_bf16_kernel(const bf16* __restrict__ X, const bf16* __restrict__ W,
                 void* __restrict__ outp, int M, int N, int K, int heads) {
  const int wave  = threadIdx.x >> 5;
  const int lane  = threadIdx.x & 31;
  const int cl    = lane & 15;
  const int hi    = lane >> 4;
  const int abase = hi * 16;
  const int bbase = hi * 16;
  const int m0 = blockIdx.x * 64 + wave * 16;
  const int n0 = blockIdx.y * 64;

  __shared__ __align__(128) bf16 wsh[2][64][64];   // 2 x 8KB

  v8f acc[4] = {};
  const bf16* ap = X + (size_t)(m0 + cl) * K + abase;

  // stage W[n0..n0+63][k..k+63] into wsh[buf]
  auto stage = [&](int buf, int k) {
#if HAVE_TDM
    if (threadIdx.x < 32)
      tdm_load_tile((unsigned)(size_t)&wsh[buf][0][0],
                    W + (size_t)n0 * K + k,
                    /*tile_k=*/64, /*tile_rows=*/64,
                    /*tens_k=*/(unsigned)(K - k),
                    /*tens_rows=*/(unsigned)(N - n0),
                    /*stride=*/(unsigned)K);
#else
    const int t = threadIdx.x, row = t >> 1, co = (t & 1) * 32;
    *(bf16x16*)&wsh[buf][row][co] =
        *(const bf16x16*)(W + (size_t)(n0 + row) * K + k + co);
    *(bf16x16*)&wsh[buf][row][co + 16] =
        *(const bf16x16*)(W + (size_t)(n0 + row) * K + k + co + 16);
#endif
  };

  stage(0, 0);
  if (threadIdx.x < 32) wait_tensorcnt0();
  __syncthreads();

  for (int k = 0; k < K; k += 64) {
    const int p = (k >> 6) & 1;
    if (k + 64 < K) stage(p ^ 1, k + 64);   // overlap DMA with compute

#pragma unroll
    for (int kk = 0; kk < 64; kk += 32) {
      bf16x16 a = load_afrag(ap + k + kk);
#pragma unroll
      for (int g = 0; g < 4; ++g) {
        bf16x16 b = *(const bf16x16*)&wsh[p][16 * g + cl][kk + bbase];
        acc[g] = wmma_bf16(a, b, acc[g]);
      }
    }

    if (threadIdx.x < 32) wait_tensorcnt0();
    __syncthreads();
  }

#pragma unroll
  for (int g = 0; g < 4; ++g) {
#pragma unroll
    for (int r = 0; r < 8; ++r) {
      const int m = m0 + r + hi * 8;
      const int n = n0 + 16 * g + cl;
      const float v = acc[g][r];
      if (MODE == 2) {
        ((float*)outp)[(size_t)m * N + n] = v;
      } else {
        const int b = m >> 11, s = m & (SEQ - 1);
        const int head = n >> 7, d = n & (HD - 1);
        bf16* o = (bf16*)outp;
        if (MODE == 0)
          o[(((size_t)b * heads + head) * SEQ + s) * HD + d] = (bf16)v;
        else
          o[(((size_t)b * heads + head) * HD + d) * SEQ + s] = (bf16)v;
      }
    }
  }
}

// ---------------------------------------------------------------------------
// RoPE (in place on bf16 [B, heads, S, hd]); position_ids is int64 [B,S]
// ---------------------------------------------------------------------------
__global__ void rope_kernel(bf16* __restrict__ buf,
                            const long long* __restrict__ pos_ids, int heads) {
  size_t idx = (size_t)blockIdx.x * blockDim.x + threadIdx.x;
  const int i = (int)(idx & 63);
  size_t t = idx >> 6;
  const int s = (int)(t & (SEQ - 1));
  t >>= 11;
  const int h = (int)(t % heads);
  const int b = (int)(t / heads);

  const float p = (float)pos_ids[(size_t)b * SEQ + s];
  const float f = p * __expf((float)i * (-9.210340371976184f / 64.0f));
  float sn, cs;
  __sincosf(f, &sn, &cs);

  const size_t base = (((size_t)b * heads + h) * SEQ + s) * HD + i;
  const float x0 = (float)buf[base];
  const float x1 = (float)buf[base + 64];
  buf[base]      = (bf16)(x0 * cs - x1 * sn);
  buf[base + 64] = (bf16)(x1 * cs + x0 * sn);
}

// ---------------------------------------------------------------------------
// Flash attention, GQA, causal. Grid (S/64, NH, B), 4 waves x 16 query rows.
// 64-wide key blocks: one max-reduction + one alpha rescale + one LDS
// round-trip per 64 keys. oacc[0..7] = O tile; oacc[8] = row-sum l via
// all-ones B fragments (matrix core computes the softmax denominator).
// ---------------------------------------------------------------------------
__global__ void __launch_bounds__(128)
flash_attn_kernel(const bf16* __restrict__ Q, const bf16* __restrict__ Kc,
                  const bf16* __restrict__ Vt, bf16* __restrict__ ctx) {
  const int wave  = threadIdx.x >> 5;
  const int lane  = threadIdx.x & 31;
  const int cl    = lane & 15;
  const int hi    = lane >> 4;
  const int abase = hi * 16;
  const int bbase = hi * 16;

  const int b   = blockIdx.z;
  const int h   = blockIdx.y;
  const int hkv = h >> 2;
  const int q0  = blockIdx.x * 64 + wave * 16;

  const bf16* qrow  = Q  + (((size_t)b * NH  + h)   * SEQ + (q0 + cl)) * HD;
  const bf16* kbase = Kc + (((size_t)b * NKV + hkv) * SEQ) * HD;
  const bf16* vbase = Vt + (((size_t)b * NKV + hkv) * HD)  * SEQ;

  bf16x16 qf[4];
#pragma unroll
  for (int f = 0; f < 4; ++f) qf[f] = load_afrag(qrow + f * 32 + abase);

  bf16x16 ones;
#pragma unroll
  for (int e = 0; e < 16; ++e) ones[e] = (bf16)1.0f;

  float mrow[8];
  v8f oacc[9] = {};    // [0..7] = O, [8] = row-sum l
#pragma unroll
  for (int r = 0; r < 8; ++r) mrow[r] = -1e30f;

  __shared__ __align__(64) bf16 pbuf[4][16][64];
  bf16(*pw)[64] = pbuf[wave];

  const float scale = 0.08838834764831845f;  // 1/sqrt(128)
  const int nkb = (q0 + 15) / 64 + 1;        // causal key-block limit

  for (int kb = 0; kb < nkb; ++kb) {
    const int k0 = kb * 64;

    // S = Q @ K^T : four 16x16 tiles over this 64-key block
    v8f sacc[4] = {};
#pragma unroll
    for (int t = 0; t < 4; ++t) {
      const bf16* kp = kbase + (size_t)(k0 + t * 16 + cl) * HD + bbase;
#pragma unroll
      for (int f = 0; f < 4; ++f) {
        bf16x16 kfrag = *(const bf16x16*)(kp + f * 32);
        sacc[t] = wmma_bf16(qf[f], kfrag, sacc[t]);
      }
    }
    if (kb + 1 < nkb)   // global_prefetch_b8 of next K block
      __builtin_prefetch(kbase + (size_t)(k0 + 64 + cl) * HD, 0, 1);

    // online softmax: one reduction + one rescale per 64 keys
#pragma unroll
    for (int r = 0; r < 8; ++r) {
      const int qi = q0 + r + hi * 8;
      float s[4];
#pragma unroll
      for (int t = 0; t < 4; ++t) {
        s[t] = sacc[t][r] * scale;
        if (k0 + t * 16 + cl > qi) s[t] = -1e30f;   // causal (== mask input)
      }
      float mx = fmaxf(fmaxf(s[0], s[1]), fmaxf(s[2], s[3]));
#pragma unroll
      for (int d = 1; d < 16; d <<= 1) mx = fmaxf(mx, __shfl_xor(mx, d, 32));
      const float mnew  = fmaxf(mrow[r], mx);
      const float alpha = __expf(mrow[r] - mnew);
      mrow[r] = mnew;
#pragma unroll
      for (int g = 0; g < 9; ++g) oacc[g][r] *= alpha;
#pragma unroll
      for (int t = 0; t < 4; ++t)
        pw[r + hi * 8][t * 16 + cl] = (bf16)__expf(s[t] - mnew);
    }

    asm volatile("s_wait_dscnt 0" ::: "memory");
    bf16x16 pf0 = load_afrag(&pw[cl][abase]);        // keys k0   .. k0+31
    bf16x16 pf1 = load_afrag(&pw[cl][32 + abase]);   // keys k0+32 .. k0+63

#pragma unroll
    for (int g = 0; g < 8; ++g) {
      const bf16* vp = vbase + (size_t)(16 * g + cl) * SEQ + k0 + bbase;
      bf16x16 vf0 = *(const bf16x16*)vp;
      bf16x16 vf1 = *(const bf16x16*)(vp + 32);
      oacc[g] = wmma_bf16(pf0, vf0, oacc[g]);
      oacc[g] = wmma_bf16(pf1, vf1, oacc[g]);
    }
    oacc[8] = wmma_bf16(pf0, ones, oacc[8]);   // l += rowsum(P)
    oacc[8] = wmma_bf16(pf1, ones, oacc[8]);
  }

#pragma unroll
  for (int r = 0; r < 8; ++r) {
    const float inv = 1.0f / oacc[8][r];       // every lane holds its row's l
    const int m = q0 + r + hi * 8;
    bf16* cp = ctx + ((size_t)b * SEQ + m) * (NH * HD) + h * HD;
#pragma unroll
    for (int g = 0; g < 8; ++g)
      cp[16 * g + cl] = (bf16)(oacc[g][r] * inv);
  }
}

// ---------------------------------------------------------------------------
// Host-side orchestration (all on `stream`, graph-capture safe)
// ---------------------------------------------------------------------------
extern "C" void kernel_launch(void* const* d_in, const int* in_sizes, int n_in,
                              void* d_out, int out_size, void* d_ws,
                              size_t ws_size, hipStream_t stream) {
  const float* x  = (const float*)d_in[0];
  const float* wq = (const float*)d_in[1];
  const float* wk = (const float*)d_in[2];
  const float* wv = (const float*)d_in[3];
  const float* wo = (const float*)d_in[4];
  // d_in[5] = attention_mask (pure causal -> synthesized inline)
  const long long* pos = (const long long*)d_in[6];
  float* out = (float*)d_out;

  char* ws = (char*)d_ws;
  bf16* xb   = (bf16*)(ws);                       // 16 MiB  [B*S, H]
  bf16* wqb  = (bf16*)(ws + 16777216);            //  8 MiB  [2048, H]
  bf16* wkb  = (bf16*)(ws + 25165824);            //  2 MiB  [512, H]
  bf16* wvb  = (bf16*)(ws + 27262976);            //  2 MiB  [512, H]
  bf16* wob  = (bf16*)(ws + 29360128);            //  8 MiB  [H, 2048]
  bf16* qb   = (bf16*)(ws + 37748736);            // 16 MiB  [B,16,S,128]
  bf16* kbuf = (bf16*)(ws + 54525952);            //  4 MiB  [B,4,S,128]
  bf16* vtb  = (bf16*)(ws + 58720256);            //  4 MiB  [B,4,128,S]
  bf16* ctxb = (bf16*)(ws + 62914560);            // 16 MiB  [B,S,2048]

  const int M = BATCH * SEQ;  // 4096

  cvt_f32_bf16<<<(8388608 + 255) / 256, 256, 0, stream>>>(x,  xb,  8388608);
  cvt_f32_bf16<<<(4194304 + 255) / 256, 256, 0, stream>>>(wq, wqb, 4194304);
  cvt_f32_bf16<<<(1048576 + 255) / 256, 256, 0, stream>>>(wk, wkb, 1048576);
  cvt_f32_bf16<<<(1048576 + 255) / 256, 256, 0, stream>>>(wv, wvb, 1048576);
  cvt_f32_bf16<<<(4194304 + 255) / 256, 256, 0, stream>>>(wo, wob, 4194304);

  gemm_bf16_kernel<0><<<dim3(M / 64, 2048 / 64), 128, 0, stream>>>(
      xb, wqb, qb, M, 2048, HID, NH);
  gemm_bf16_kernel<0><<<dim3(M / 64, 512 / 64), 128, 0, stream>>>(
      xb, wkb, kbuf, M, 512, HID, NKV);
  gemm_bf16_kernel<1><<<dim3(M / 64, 512 / 64), 128, 0, stream>>>(
      xb, wvb, vtb, M, 512, HID, NKV);

  rope_kernel<<<(BATCH * NH * SEQ * 64) / 256, 256, 0, stream>>>(qb, pos, NH);
  rope_kernel<<<(BATCH * NKV * SEQ * 64) / 256, 256, 0, stream>>>(kbuf, pos, NKV);

  flash_attn_kernel<<<dim3(SEQ / 64, NH, BATCH), 128, 0, stream>>>(
      qb, kbuf, vtb, ctxb);

  gemm_bf16_kernel<2><<<dim3(M / 64, 2048 / 64), 128, 0, stream>>>(
      ctxb, wob, out, M, 2048, HID, 0);

  (void)in_sizes; (void)n_in; (void)out_size; (void)ws_size;
}
